// HomoRegression_80075370266813
// MI455X (gfx1250) — matrix operand order repaired and checked by reading, not compiled, not run on hardware
//
#include <hip/hip_runtime.h>
#include <hip/hip_bf16.h>

// ---------------------------------------------------------------------------
// Fused edge-MLP link predictor for MI455X (gfx1250, wave32, WMMA).
//   out[e] = sigmoid( W2 . relu( W1 @ concat(x[src[e]], x[dst[e]]) + b1 ) + b2 )
//
// v4: double-buffered gather WITHOUT manual interleave (v3 blew past 256
//     VGPRs -> s_set_vgpr_msb storm + 5x code bloat).
//   * Even N-tile B fragments hoisted into 128 VGPRs straight from GLOBAL W1
//     (read once per block); LDS holds only the 8 odd N-tiles (128KB).
//   * Ping-pong 48-edge A tiles (2 x 48KB). Staging of the next tile is a
//     COMPACT rolled loop before the K loop with no intervening barrier:
//     waves drift between phases, so gather latency of one wave hides under
//     the co-resident wave's WMMA issue. One end-of-tile barrier certifies
//     both "next buffer staged" and "current buffer free".
//   * Branchless gather chunk (clamp + cndmask) -> no save-exec chains.
//   * Steady-state K loop: 8 ds_load_b128 per 6 v_wmma_f32_16x16x32_bf16.
// ---------------------------------------------------------------------------

typedef __attribute__((ext_vector_type(16))) __bf16 v16bf;
typedef __attribute__((ext_vector_type(8)))  __bf16 v8bf;
typedef __attribute__((ext_vector_type(8)))  float  v8f;

#define HID      256
#define K2       512            // 2*HID
#define KSTEPS   16             // K2 / 32
#define TILE_M   48             // edges per tile (3 WMMA M-tiles)
#define NTHREADS 256            // 8 wave32s
#define ASZ      (TILE_M * KSTEPS * 32)   // A buffer elements (24576)

__device__ __forceinline__ unsigned short f2bf_bits(float f) {
    unsigned u = __float_as_uint(f);
    unsigned r = u + 0x7FFFu + ((u >> 16) & 1u);   // round-to-nearest-even
    return (unsigned short)(r >> 16);
}
__device__ __forceinline__ unsigned pack2bf(float a, float b) {
    return (unsigned)f2bf_bits(a) | ((unsigned)f2bf_bits(b) << 16);
}
__device__ __forceinline__ v16bf cat16(v8bf lo, v8bf hi) {
    return __builtin_shufflevector(lo, hi, 0,1,2,3,4,5,6,7,8,9,10,11,12,13,14,15);
}

// Branchless gather+convert of one 4-float chunk into a fragment-major A
// buffer: lane(g,ln) of frag (mt,s) holds K = s*32+g*16+{0..15} in j=0..15.
__device__ __forceinline__ void stage_chunk(int c, __bf16* __restrict__ dst,
                                            int ebase, int E,
                                            const float* __restrict__ x,
                                            const int* __restrict__ ei) {
    const int row = c >> 7;                 // edge row in tile
    const int k   = (c & 127) << 2;         // K col (multiple of 4)
    const int sel = k >> 8;                 // 0: src endpoint, 1: dst endpoint
    int e = ebase + row;
    e = (e < E) ? e : (E - 1);              // clamp tail (output is guarded)
    const int node = ei[(sel ? E : 0) + e];
    const int off  = k - (sel << 8);
    const float4 v = *(const float4*)(x + (size_t)node * HID + off);
    const int mt  = row >> 4;
    const int s   = k >> 5;
    const int rem = k & 31;
    const int L   = ((rem >> 4) & 1) * 16 + (row & 15);
    const int j   = rem & 15;
    uint2 p; p.x = pack2bf(v.x, v.y); p.y = pack2bf(v.z, v.w);
    *(uint2*)(dst + ((mt * KSTEPS + s) * 32 + L) * 16 + j) = p;
}

__global__ __launch_bounds__(NTHREADS) void
edge_mlp_wmma_kernel(const float* __restrict__ x,
                     const int*   __restrict__ ei,     // [2, E]
                     const float* __restrict__ W1,     // [256, 512] row-major
                     const float* __restrict__ b1,     // [256]
                     const float* __restrict__ W2,     // [256]
                     const float* __restrict__ b2,     // [1]
                     float*       __restrict__ out,    // [E]
                     int E, int nTiles)
{
    __shared__ alignas(32) __bf16 W1s[KSTEPS * 8 * 32 * 16]; // odd N-tiles, 131,072 B
    __shared__ alignas(32) __bf16 As[2 * ASZ];               // ping-pong,    98,304 B
    __shared__ float partialS[TILE_M];                       //                  192 B

    const int tid  = threadIdx.x;
    const int wave = tid >> 5;          // 0..7 -> owns N cols [wave*32, wave*32+32)
    const int lane = tid & 31;
    const int g    = lane >> 4;
    const int ln   = lane & 15;

    // ---- stage ODD N-tiles of W1 (fp32 -> bf16 LDS, fragment-major) ----
    for (int c = tid; c < 128 * 128; c += NTHREADS) {
        const int rl = c >> 7;                 // 0..127 local odd row
        const int k  = (c & 127) << 2;
        const int tIdx = rl >> 4;              // 0..7
        const int nrow = (2 * tIdx + 1) * 16 + (rl & 15);
        const float4 v = *(const float4*)(W1 + nrow * K2 + k);
        const int s = k >> 5;
        const int L = ((k >> 4) & 1) * 16 + (rl & 15);
        const int j = k & 15;
        uint2 p; p.x = pack2bf(v.x, v.y); p.y = pack2bf(v.z, v.w);
        *(uint2*)(&W1s[((s * 8 + tIdx) * 32 + L) * 16 + j]) = p;
    }

    // ---- hoist EVEN N-tile B fragments from GLOBAL (once/block): 128 VGPRs
    //      lane(g,ln) of step s holds W1[2*wave*16+ln][s*32+g*16+{0..15}] ----
    v16bf Bh[KSTEPS];
    {
        const float* w1row = W1 + (size_t)(2 * wave * 16 + ln) * K2;
#pragma unroll
        for (int s = 0; s < KSTEPS; ++s) {
            const int kb = s * 32 + g * 16;
            v16bf bb;
#pragma unroll
            for (int q = 0; q < 4; ++q) {
                const float4 v = *(const float4*)(w1row + kb + q * 4);
                bb[q * 4 + 0] = __builtin_bit_cast(__bf16, f2bf_bits(v.x));
                bb[q * 4 + 1] = __builtin_bit_cast(__bf16, f2bf_bits(v.y));
                bb[q * 4 + 2] = __builtin_bit_cast(__bf16, f2bf_bits(v.z));
                bb[q * 4 + 3] = __builtin_bit_cast(__bf16, f2bf_bits(v.w));
            }
            Bh[s] = bb;
        }
    }

    // ---- per-lane fc2 constants ----
    float b1v[2], w2v[2];
#pragma unroll
    for (int t = 0; t < 2; ++t) {
        const int n = wave * 32 + t * 16 + ln;
        b1v[t] = b1[n];
        w2v[t] = W2[n];
    }
    const float b2v = b2[0];

    // ---- prologue: stage first tile into buffer 0 ----
    for (int c = tid; c < TILE_M * K2 / 4; c += NTHREADS)
        stage_chunk(c, As, blockIdx.x * TILE_M, E, x, ei);
    if (tid < TILE_M) partialS[tid] = b2v;
    __syncthreads();

    const __bf16* curA = As;
    __bf16*       nxtA = As + ASZ;

    for (int tile = blockIdx.x; tile < nTiles; tile += gridDim.x) {
        // ---- stage next tile (compact loop, NO barrier before compute:
        //      waves drift, gather hides under other waves' WMMA) ----
        const int tileNext = tile + gridDim.x;
        if (tileNext < nTiles) {
            const int ebaseN = tileNext * TILE_M;
            for (int c = tid; c < TILE_M * K2 / 4; c += NTHREADS)
                stage_chunk(c, nxtA, ebaseN, E, x, ei);
        }

        // ---- K loop on current buffer: 8 ds_load_b128 + 6 WMMA per step ----
        v8f acc[3][2] = {};
#pragma unroll
        for (int s = 0; s < KSTEPS; ++s) {
            const __bf16* pb = &W1s[((s * 8 + wave) * 32 + lane) * 16];
            const v16bf bOdd = cat16(*(const v8bf*)pb, *(const v8bf*)(pb + 8));
#pragma unroll
            for (int mt = 0; mt < 3; ++mt) {
                const __bf16* pa = curA + ((mt * KSTEPS + s) * 32 + lane) * 16;
                const v16bf a = cat16(*(const v8bf*)pa, *(const v8bf*)(pa + 8));
                acc[mt][0] = __builtin_amdgcn_wmma_f32_16x16x32_bf16(
                                false, a, false, Bh[s], (short)0, acc[mt][0], false, false);
                acc[mt][1] = __builtin_amdgcn_wmma_f32_16x16x32_bf16(
                                false, a, false, bOdd,  (short)0, acc[mt][1], false, false);
            }
        }

        // ---- fused epilogue: h = relu(acc + b1); partial = h . W2 slice ----
        // acc[mt][t][r] at lane(g,ln) is h[m = mt*16 + 8g + r][n = wave*32 + t*16 + ln]
        float p[3][8];
#pragma unroll
        for (int mt = 0; mt < 3; ++mt)
#pragma unroll
            for (int r = 0; r < 8; ++r) {
                float s0 = fmaxf(acc[mt][0][r] + b1v[0], 0.f) * w2v[0];
                p[mt][r] = fmaf(fmaxf(acc[mt][1][r] + b1v[1], 0.f), w2v[1], s0);
            }
#pragma unroll
        for (int off = 1; off < 16; off <<= 1)
#pragma unroll
            for (int mt = 0; mt < 3; ++mt)
#pragma unroll
                for (int r = 0; r < 8; ++r)
                    p[mt][r] += __shfl_xor(p[mt][r], off, 32);
        if (ln == 0) {
#pragma unroll
            for (int mt = 0; mt < 3; ++mt)
#pragma unroll
                for (int r = 0; r < 8; ++r)
                    atomicAdd(&partialS[mt * 16 + g * 8 + r], p[mt][r]); // ds_add_f32
        }
        __syncthreads();   // partials done; next buffer staged; cur free

        if (tid < TILE_M) {
            const int e = tile * TILE_M + tid;
            if (e < E) out[e] = 1.f / (1.f + __expf(-partialS[tid]));
            partialS[tid] = b2v;          // re-armed by its own reader
        }
        __syncthreads();

        // swap ping-pong buffers
        __bf16* t = (__bf16*)curA; curA = nxtA; nxtA = t;
    }
}

extern "C" void kernel_launch(void* const* d_in, const int* in_sizes, int n_in,
                              void* d_out, int out_size, void* d_ws, size_t ws_size,
                              hipStream_t stream) {
    const float* x  = (const float*)d_in[0];
    const int*   ei = (const int*)  d_in[1];   // [2, E]
    const float* W1 = (const float*)d_in[2];
    const float* b1 = (const float*)d_in[3];
    const float* W2 = (const float*)d_in[4];
    const float* b2 = (const float*)d_in[5];
    float* out = (float*)d_out;

    const int E = in_sizes[1] / 2;
    const int nTiles = (E + TILE_M - 1) / TILE_M;
    int blocks = nTiles < 640 ? nTiles : 640;   // persistent, ~1 block per WGP
    if (blocks < 1) blocks = 1;

    edge_mlp_wmma_kernel<<<blocks, NTHREADS, 0, stream>>>(
        x, ei, W1, b1, W2, b2, out, E, nTiles);
}
